// FusedMoE_50603304681702
// MI455X (gfx1250) — matrix operand order repaired and checked by reading, not compiled, not run on hardware
//
#include <hip/hip_runtime.h>
#include <hip/hip_bf16.h>
#include <math.h>

// ---- problem constants (from reference) ----
constexpr int En   = 8;
constexpr int Tn   = 512;
constexpr int Hn   = 1024;
constexpr int Fn   = 2816;
constexpr int Gn   = 128;
constexpr int N2F  = 2 * Fn;      // 5632
constexpr int TOPK = 2;
constexpr int KS   = 64;          // K-slab per barrier (2 WMMA K-steps)

typedef __attribute__((ext_vector_type(16))) _Float16 v16h;
typedef __attribute__((ext_vector_type(8)))  _Float16 v8h;
typedef __attribute__((ext_vector_type(8)))  float    v8f;

// ---------------------------------------------------------------------------
// async global->LDS (CDNA5 GLOBAL_LOAD_ASYNC_TO_LDS_B128, ASYNCcnt-tracked)
// ---------------------------------------------------------------------------
__device__ __forceinline__ void async_copy_b128(const void* gaddr, void* lds_ptr) {
    unsigned lds_off = (unsigned)(unsigned long long)(uintptr_t)lds_ptr; // addr[31:0] = LDS offset
    asm volatile("global_load_async_to_lds_b128 %0, %1, off"
                 :
                 : "v"(lds_off), "v"((unsigned long long)(uintptr_t)gaddr)
                 : "memory");
}
__device__ __forceinline__ void wait_asynccnt0() {
    asm volatile("s_wait_asynccnt 0" ::: "memory");
}

// ---------------------------------------------------------------------------
// Kernel 1: router softmax + top-2 (renormalized)
// ---------------------------------------------------------------------------
__global__ void moe_router(const float* __restrict__ logits,
                           float* __restrict__ topk_w,
                           int* __restrict__ topk_id) {
    int t = blockIdx.x * blockDim.x + threadIdx.x;
    if (t >= Tn) return;
    float v[En];
    float mx = -3.4e38f;
    for (int e = 0; e < En; ++e) { v[e] = logits[t * En + e]; mx = fmaxf(mx, v[e]); }
    for (int e = 0; e < En; ++e) v[e] = __expf(v[e] - mx);
    int i0 = 0;
    for (int e = 1; e < En; ++e) if (v[e] > v[i0]) i0 = e;
    int i1 = (i0 == 0) ? 1 : 0;
    for (int e = 0; e < En; ++e) if (e != i0 && v[e] > v[i1]) i1 = e;
    float w0 = v[i0], w1 = v[i1];
    float inv = 1.0f / (w0 + w1);
    topk_id[t * 2 + 0] = i0; topk_id[t * 2 + 1] = i1;
    topk_w [t * 2 + 0] = w0 * inv; topk_w[t * 2 + 1] = w1 * inv;
}

// ---------------------------------------------------------------------------
// Kernel 2: deterministic per-expert token lists + prefix offsets + slot map
// ---------------------------------------------------------------------------
__global__ void moe_build_lists(const int* __restrict__ topk_id,
                                int* __restrict__ counts,
                                int* __restrict__ row_base,
                                int* __restrict__ tok_list,
                                int* __restrict__ slot_map) {
    __shared__ int rb_s[En + 1];
    int e = threadIdx.x;
    if (e < En) {
        int c = 0;
        for (int i = 0; i < Tn * TOPK; ++i) c += (topk_id[i] == e);
        counts[e] = c;
    }
    __syncthreads();
    if (threadIdx.x == 0) {
        int acc = 0;
        for (int j = 0; j < En; ++j) { rb_s[j] = acc; row_base[j] = acc; acc += counts[j]; }
        rb_s[En] = acc;
    }
    __syncthreads();
    if (e < En) {
        int pos = 0, base = rb_s[e];
        for (int t = 0; t < Tn; ++t)
            for (int k = 0; k < TOPK; ++k)
                if (topk_id[t * 2 + k] == e) {
                    tok_list[e * Tn + pos] = t;
                    slot_map[t * 2 + k] = base + pos;
                    ++pos;
                }
    }
}

// ---------------------------------------------------------------------------
// Fragment helpers (ISA 16x16x32 f16 WMMA layouts, wave32)
// ---------------------------------------------------------------------------
__device__ __forceinline__ v16h load_A_frag(const _Float16* s, int base) {
    // A 16x32: M = lane&15, lane half (lane>>4) selects K offsets {0..7,16..23} / {8..15,24..31}
    v8h lo = *(const v8h*)(s + base);
    v8h hi = *(const v8h*)(s + base + 16);
    return __builtin_shufflevector(lo, hi, 0,1,2,3,4,5,6,7,8,9,10,11,12,13,14,15);
}
__device__ __forceinline__ v16h load_B_frag(const _Float16* s, int krow, int ncol) {
    // B 32x16: N = lane&15, lane half selects K offsets {0..7,16..23} / {8..15,24..31} (pitch 32)
    v16h r;
    #pragma unroll
    for (int i = 0; i < 8; ++i) {
        r[i]     = s[(krow + i) * 32 + ncol];
        r[8 + i] = s[(krow + 16 + i) * 32 + ncol];
    }
    return r;
}

// ---------------------------------------------------------------------------
// Kernel 3: gathered GEMM1 + fused SiLU*up -> f16 activation workspace
// block = 256 threads (8 waves): tile M=64 x (N=32 gate + N=32 up), K-slab 64
// ---------------------------------------------------------------------------
__global__ __launch_bounds__(256) void moe_gemm1(
        const float* __restrict__ x,
        const int*   __restrict__ w13_q,
        const float* __restrict__ w13_s,
        const int*   __restrict__ counts,
        const int*   __restrict__ row_base,
        const int*   __restrict__ tok_list,
        _Float16*    __restrict__ a_ws) {
    const int e  = blockIdx.x >> 3;
    const int m0 = (blockIdx.x & 7) * 64;
    const int cnt = counts[e];
    if (m0 >= cnt) return;
    const int n0 = blockIdx.y * 32;               // over F (gate half)

    __shared__ alignas(16) _Float16 xs[64 * KS];  // 8 KB
    __shared__ alignas(16) _Float16 bgs[KS * 32]; // 4 KB
    __shared__ alignas(16) _Float16 bus[KS * 32]; // 4 KB

    const int tid = threadIdx.x;

    // x staging: 16 f16 per thread (4 x float4 loads)
    const int xr = tid >> 2, xc0 = (tid & 3) * 16;
    const int tok = (m0 + xr < cnt) ? tok_list[e * Tn + m0 + xr] : -1;
    if (tok < 0) {                                 // padding rows: zero once
        _Float16* d = &xs[xr * KS + xc0];
        for (int j = 0; j < 16; ++j) d[j] = (_Float16)0.0f;
    }

    // cooperative int4 dequant: 2 packed dwords per thread (kw = dkw, dkw+4)
    const bool isU = tid >= 128;
    const int dn  = tid & 31;
    const int dkw = (tid >> 5) & 3;
    const int ncol = n0 + dn + (isU ? Fn : 0);
    const int*   qbase = w13_q + (size_t)e * (Hn / 8) * N2F + ncol;
    const float* sbase = w13_s + (size_t)e * (Hn / Gn) * N2F + ncol;
    _Float16* bptr = isU ? bus : bgs;

    const int lane = tid & 31, wid = tid >> 5;
    const int msub = wid & 3, nsub = wid >> 2;    // 4 M-subtiles x 2 N-subtiles
    const int arow = (msub * 16 + (lane & 15)) * KS + (lane >> 4) * 8;

    v8f accg = {}; v8f accu = {};

    for (int k0 = 0; k0 < Hn; k0 += KS) {
        if (tok >= 0) {
            const float4* xp = (const float4*)(x + (size_t)tok * Hn + k0 + xc0);
            _Float16* d = &xs[xr * KS + xc0];
            #pragma unroll
            for (int q = 0; q < 4; ++q) {
                float4 p = xp[q];
                d[q*4+0]=(_Float16)p.x; d[q*4+1]=(_Float16)p.y;
                d[q*4+2]=(_Float16)p.z; d[q*4+3]=(_Float16)p.w;
            }
        }
        unsigned w0 = (unsigned)qbase[((k0 >> 3) + dkw)     * N2F];
        unsigned w1 = (unsigned)qbase[((k0 >> 3) + dkw + 4) * N2F];
        float sc = sbase[(k0 >> 7) * N2F];         // 64-aligned slab stays in one G=128 group
        #pragma unroll
        for (int p = 0; p < 8; ++p) {
            bptr[((dkw    ) * 8 + p) * 32 + dn] =
                (_Float16)((float)((int)((w0 >> (4 * p)) & 15u) - 8) * sc);
            bptr[((dkw + 4) * 8 + p) * 32 + dn] =
                (_Float16)((float)((int)((w1 >> (4 * p)) & 15u) - 8) * sc);
        }
        if (k0 + KS < Hn)
            __builtin_prefetch(qbase + (((k0 + KS) >> 3) + dkw) * N2F, 0, 1);
        __syncthreads();

        #pragma unroll
        for (int s = 0; s < 2; ++s) {             // two K=32 WMMA steps per slab
            v16h A  = load_A_frag(xs, arow + s * 32);
            v16h Bg = load_B_frag(bgs, s * 32 + (lane >> 4) * 8, nsub * 16 + (lane & 15));
            v16h Bu = load_B_frag(bus, s * 32 + (lane >> 4) * 8, nsub * 16 + (lane & 15));
            accg = __builtin_amdgcn_wmma_f32_16x16x32_f16(false, A, false, Bg, (short)0, accg, false, false);
            accu = __builtin_amdgcn_wmma_f32_16x16x32_f16(false, A, false, Bu, (short)0, accu, false, false);
        }
        __syncthreads();
    }

    // fused SiLU(gate) * up epilogue -> compact f16 activation
    const int rb = row_base[e];
    const int half = lane >> 4;
    const int col = n0 + nsub * 16 + (lane & 15);
    #pragma unroll
    for (int r = 0; r < 8; ++r) {
        int mloc = msub * 16 + r + 8 * half;      // C/D layout: M = r + 8*(lane>>4)
        if (m0 + mloc >= cnt) continue;
        float g = accg[r], u = accu[r];
        float a = (g / (1.0f + __expf(-g))) * u;
        a_ws[(size_t)(rb + m0 + mloc) * Fn + col] = (_Float16)a;
    }
}

// ---------------------------------------------------------------------------
// Kernel 4: GEMM2 (activation @ dequant W2) -> unweighted y rows
// activation staged with GLOBAL_LOAD_ASYNC_TO_LDS_B128 (ASYNCcnt path)
// ---------------------------------------------------------------------------
__global__ __launch_bounds__(256) void moe_gemm2(
        const _Float16* __restrict__ a_ws,
        const int*      __restrict__ w2_q,
        const float*    __restrict__ w2_s,
        const int*      __restrict__ counts,
        const int*      __restrict__ row_base,
        float*          __restrict__ y_ws) {
    const int e  = blockIdx.x >> 3;
    const int m0 = (blockIdx.x & 7) * 64;
    const int cnt = counts[e];
    if (m0 >= cnt) return;
    const int n0 = blockIdx.y * 32;               // over H
    const int rb = row_base[e];

    __shared__ alignas(16) _Float16 as[64 * KS];  // 8 KB
    __shared__ alignas(16) _Float16 bs[KS * 32];  // 4 KB

    const int tid = threadIdx.x;
    const int ar = tid >> 2, ac0 = (tid & 3) * 16;
    const bool avalid = (m0 + ar < cnt);
    const _Float16* abase = a_ws + (size_t)(rb + m0 + ar) * Fn + ac0;
    _Float16* adst = &as[ar * KS + ac0];
    if (!avalid) {                                 // padding rows: zero once
        for (int j = 0; j < 16; ++j) adst[j] = (_Float16)0.0f;
    }

    // dequant: one packed dword (8 nibbles) per thread, 8 kw-rows x 32 n
    const int dn  = tid & 31;
    const int dkw = tid >> 5;                     // 0..7
    const int*   qbase = w2_q + (size_t)e * (Fn / 8) * Hn + (n0 + dn);
    const float* sbase = w2_s + (size_t)e * (Fn / Gn) * Hn + (n0 + dn);

    const int lane = tid & 31, wid = tid >> 5;
    const int msub = wid & 3, nsub = wid >> 2;
    const int arow = (msub * 16 + (lane & 15)) * KS + (lane >> 4) * 8;

    v8f acc = {};

    for (int k0 = 0; k0 < Fn; k0 += KS) {
        if (avalid) {                              // async f16 global -> LDS, no VGPR bounce
            async_copy_b128(abase + k0,     adst);
            async_copy_b128(abase + k0 + 8, adst + 8);
        }
        unsigned w = (unsigned)qbase[((k0 >> 3) + dkw) * Hn];
        float sc = sbase[(k0 >> 7) * Hn];
        #pragma unroll
        for (int p = 0; p < 8; ++p)
            bs[(dkw * 8 + p) * 32 + dn] =
                (_Float16)((float)((int)((w >> (4 * p)) & 15u) - 8) * sc);
        if (k0 + KS < Fn)
            __builtin_prefetch(qbase + (((k0 + KS) >> 3) + dkw) * Hn, 0, 1);
        wait_asynccnt0();
        __syncthreads();

        #pragma unroll
        for (int s = 0; s < 2; ++s) {
            v16h A = load_A_frag(as, arow + s * 32);
            v16h B = load_B_frag(bs, s * 32 + (lane >> 4) * 8, nsub * 16 + (lane & 15));
            acc = __builtin_amdgcn_wmma_f32_16x16x32_f16(false, A, false, B, (short)0, acc, false, false);
        }
        __syncthreads();
    }

    const int half = lane >> 4;
    const int col = n0 + nsub * 16 + (lane & 15);
    #pragma unroll
    for (int r = 0; r < 8; ++r) {
        int mloc = msub * 16 + r + 8 * half;
        if (m0 + mloc >= cnt) continue;
        y_ws[(size_t)(rb + m0 + mloc) * Hn + col] = acc[r];
    }
}

// ---------------------------------------------------------------------------
// Kernel 5: combine: out[t,h] = w0*y[slot0,h] + w1*y[slot1,h]  (no atomics)
// ---------------------------------------------------------------------------
__global__ void moe_combine(const float* __restrict__ y_ws,
                            const float* __restrict__ topk_w,
                            const int*   __restrict__ slot_map,
                            float*       __restrict__ out) {
    int idx = (blockIdx.x * blockDim.x + threadIdx.x) * 4;
    if (idx >= Tn * Hn) return;
    int t = idx / Hn, h = idx % Hn;
    float w0 = topk_w[t * 2], w1 = topk_w[t * 2 + 1];
    int   s0 = slot_map[t * 2], s1 = slot_map[t * 2 + 1];
    float4 y0 = *(const float4*)(y_ws + (size_t)s0 * Hn + h);
    float4 y1 = *(const float4*)(y_ws + (size_t)s1 * Hn + h);
    float4 o;
    o.x = w0 * y0.x + w1 * y1.x;
    o.y = w0 * y0.y + w1 * y1.y;
    o.z = w0 * y0.z + w1 * y1.z;
    o.w = w0 * y0.w + w1 * y1.w;
    *(float4*)(out + idx) = o;
}

// ---------------------------------------------------------------------------
// host launcher
// ---------------------------------------------------------------------------
extern "C" void kernel_launch(void* const* d_in, const int* in_sizes, int n_in,
                              void* d_out, int out_size, void* d_ws, size_t ws_size,
                              hipStream_t stream) {
    const float* x       = (const float*)d_in[0];
    const float* logits  = (const float*)d_in[1];
    const int*   w13_q   = (const int*)  d_in[2];
    const int*   w2_q    = (const int*)  d_in[3];
    const float* w13_s   = (const float*)d_in[4];
    const float* w2_s    = (const float*)d_in[5];
    float*       out     = (float*)d_out;

    // workspace layout
    char* ws = (char*)d_ws;
    int*       topk_id  = (int*)  (ws + 0);          //  4 KB
    float*     topk_w   = (float*)(ws + 4096);       //  4 KB
    int*       counts   = (int*)  (ws + 8192);       //  256 B slot
    int*       row_base = (int*)  (ws + 8448);       //  256 B slot
    int*       tok_list = (int*)  (ws + 8704);       // 16 KB
    int*       slot_map = (int*)  (ws + 8704 + 16384);          // 4 KB
    _Float16*  a_ws     = (_Float16*)(ws + 32768);              // 1024*2816*2 = 5767168 B
    float*     y_ws     = (float*)(ws + 32768 + 5767168);       // 1024*1024*4 = 4194304 B

    moe_router<<<dim3((Tn + 127) / 128), dim3(128), 0, stream>>>(logits, topk_w, topk_id);
    moe_build_lists<<<dim3(1), dim3(32), 0, stream>>>(topk_id, counts, row_base, tok_list, slot_map);
    moe_gemm1<<<dim3(En * 8, Fn / 32), dim3(256), 0, stream>>>(
        x, w13_q, w13_s, counts, row_base, tok_list, a_ws);
    moe_gemm2<<<dim3(En * 8, Hn / 32), dim3(256), 0, stream>>>(
        a_ws, w2_q, w2_s, counts, row_base, y_ws);
    moe_combine<<<dim3((Tn * Hn / 4 + 255) / 256), dim3(256), 0, stream>>>(
        y_ws, topk_w, slot_map, out);
}
